// MP_Layer_1099511627812
// MI455X (gfx1250) — compile-verified, matrix operand order checked by reading, not run on hardware
//
#include <hip/hip_runtime.h>
#include <math.h>

// Shapes fixed by the reference: B=8, N=64, D=64, E=32
#define B_ 8
#define N_ 64
#define D_ 64
#define E_ 32

typedef __attribute__((ext_vector_type(16))) _Float16 v16h;
typedef __attribute__((ext_vector_type(8)))  float    v8f;
typedef __attribute__((ext_vector_type(4)))  float    v4f;

// ---------------------------------------------------------------------------
// Prep: swizzle W_e (E x D*D, f32) into f16 WMMA B-operand tiles.
// Tile w = fi*4 + fjb covers cols [fi*64 + fjb*16, +16), K = 0..31.
// Per-lane layout (B is K x N, lane = col n = lane%16, K split by lane half):
//   kb = (lane>=16)*8 ; h[i] = K = (i<8 ? kb+i : 16+kb+(i-8))
// ---------------------------------------------------------------------------
__global__ __launch_bounds__(32) void prepW_kernel(const float* __restrict__ W_e,
                                                   _Float16* __restrict__ Wh) {
    const int w    = blockIdx.x;          // 0..255
    const int lane = threadIdx.x;         // 0..31
    const int fi   = w >> 2;
    const int fjb  = w & 3;
    const int n    = lane & 15;
    const int kb   = (lane >> 4) * 8;
    _Float16* outp = Wh + (((size_t)w * 32) + lane) * 16;
    const int col  = fi * D_ + fjb * 16 + n;
#pragma unroll
    for (int i = 0; i < 16; ++i) {
        const int K = (i < 8) ? (kb + i) : (16 + kb + (i - 8));
        outp[i] = (_Float16)W_e[(size_t)K * (D_ * D_) + col];
    }
}

// ---------------------------------------------------------------------------
// Fused message kernel: one workgroup per (b, u) target node, 4 waves.
// Wave w computes agg[b, u, fi] for fi in [w*16, w*16+16).
// 16 tiles per fi flow through a 2-deep pipeline (WMMA(t) ; epilogue(t-1))
// so the relu+fma epilogue co-executes with the matrix pipe. The epilogue
// is fully IR-visible (add + max + fma, all dual-issuable) so the compiler
// enforces the CDNA5 WMMA->VALU hazard rules.
// ---------------------------------------------------------------------------
__global__ __launch_bounds__(128) void msg_kernel(const float* __restrict__ nodes,
                                                  const float* __restrict__ edges,
                                                  const float* __restrict__ mask,
                                                  const float* __restrict__ b_e,
                                                  const _Float16* __restrict__ Wh,
                                                  float* __restrict__ agg) {
    __shared__ float xnode[N_ * D_];                 // 16 KB: nodes[b]
    const int b   = blockIdx.x >> 6;                 // 512 blocks = (b,u)
    const int u   = blockIdx.x & 63;
    const int tid = threadIdx.x;

    // vectorized stage of nodes[b] into LDS
    {
        const v4f* src = (const v4f*)(nodes + (size_t)b * N_ * D_);
        v4f* dst = (v4f*)xnode;
        for (int i = tid; i < (N_ * D_) / 4; i += 128) dst[i] = src[i];
    }
    __syncthreads();

    const int wave = tid >> 5;
    const int lane = tid & 31;
    const int ln16 = lane & 15;
    const int hi   = lane >> 4;                      // lane half
    const int kb   = hi * 8;

    // ---- A-operand tiles: 4 tiles of 16 edges (sources v), K = E = 32 ----
    v16h at[4];
#pragma unroll
    for (int vb = 0; vb < 4; ++vb) {
        const float* ep =
            edges + ((size_t)b * N_ * N_ + u * N_ + vb * 16 + ln16) * E_;
#pragma unroll
        for (int i = 0; i < 16; ++i) {
            const int K = (i < 8) ? (kb + i) : (16 + kb + (i - 8));
            at[vb][i] = (_Float16)ep[K];
        }
    }

    // ---- Hoisted weight tiles xm[v,fj] = mask[b,u*64+v] * nodes[b,v,fj] ----
    // C/D element (VGPR r, this lane) = (row r + 8*hi, col ln16) of the tile.
    const float* mrow = mask + (size_t)b * N_ * N_ + u * N_;
    float xm[16][8];
#pragma unroll
    for (int vb = 0; vb < 4; ++vb)
#pragma unroll
        for (int fjb = 0; fjb < 4; ++fjb)
#pragma unroll
            for (int r = 0; r < 8; ++r) {
                const int v  = vb * 16 + r + hi * 8;
                const int fj = fjb * 16 + ln16;
                xm[vb * 4 + fjb][r] = mrow[v] * xnode[v * D_ + fj];
            }

    float* aggrow = agg + ((size_t)b * N_ + u) * D_;
    const int fi0 = wave * 16;

// tile t (0..15): fjb = t>>2, vb = t&3 ; xm tile index = vb*4 + fjb
#define LOADB(f) (*(const v16h*)(Wh + (((size_t)(fi * 4 + (f))) * 32 + lane) * 16))
#define WM(vb_, fh_) __builtin_amdgcn_wmma_f32_16x16x32_f16( \
        false, at[(vb_)], false, bt[(fh_)], (short)0, czero, false, false)
// epilogue: z = d + bias (fadd => canonical => single v_max), relu, fma
#define EPI(dreg_, tt_)                                                     \
    {                                                                       \
        _Pragma("unroll")                                                   \
        for (int r = 0; r < 8; ++r)                                         \
            acc[r] += fmaxf((dreg_)[r] + bev[(tt_) >> 2], 0.f) *            \
                      xm[((tt_) & 3) * 4 + ((tt_) >> 2)][r];                \
    }

    for (int fi = fi0; fi < fi0 + 16; ++fi) {
        float acc[8];
#pragma unroll
        for (int r = 0; r < 8; ++r) acc[r] = 0.f;

        // per-column bias values for this fi (added in the epilogue)
        float bev[4];
#pragma unroll
        for (int j = 0; j < 4; ++j) bev[j] = b_e[fi * D_ + j * 16 + ln16];

        const v8f czero = {};
        v16h bt[2];   // rotating B tiles (per fjb)
        v8f  db[2];   // rotating WMMA destinations

        // ---- pipeline prologue ----
        bt[0] = LOADB(0);
        db[0] = WM(0, 0);                       // W0 (tile 0: fjb0, vb0)
        db[1] = WM(1, 0);                       // W1 (tile 1: fjb0, vb1)
        bt[1] = LOADB(1);                       // prefetch fjb1 B tile
        EPI(db[0], 0);                          // E0

        // ---- steady state: W_t ; (prefetch) ; E_{t-1} ----
#pragma unroll
        for (int t = 2; t <= 15; ++t) {
            const int f  = t >> 2;
            const int vb = t & 3;
            db[t & 1] = WM(vb, f & 1);
            if (vb == 0 && f < 3)               // prefetch next fjb, 4 tiles ahead
                bt[(f + 1) & 1] = LOADB(f + 1);
            EPI(db[(t - 1) & 1], t - 1);
        }
        EPI(db[1], 15);                         // tail: E15

        // tree-sum the 8 independent chains, then full 32-lane reduction
        float accf = ((acc[0] + acc[1]) + (acc[2] + acc[3])) +
                     ((acc[4] + acc[5]) + (acc[6] + acc[7]));
#pragma unroll
        for (int off = 16; off > 0; off >>= 1)
            accf += __shfl_xor(accf, off, 32);
        if (lane == 0) aggrow[fi] = accf;
    }
#undef LOADB
#undef WM
#undef EPI
}

// ---------------------------------------------------------------------------
// GRU step 1: h1 = GRU(nodes, h=0). One block (64 threads) per row.
// ---------------------------------------------------------------------------
__device__ __forceinline__ float sigmoidf_(float x) { return 1.f / (1.f + expf(-x)); }

__global__ __launch_bounds__(64) void gru1_kernel(const float* __restrict__ x_in,
                                                  const float* __restrict__ Kk,
                                                  const float* __restrict__ bias,
                                                  float* __restrict__ h1) {
    __shared__ float xr[D_];
    const int row = blockIdx.x, d = threadIdx.x;
    xr[d] = x_in[(size_t)row * D_ + d];
    __syncthreads();
    float sz = bias[d], sr = bias[D_ + d], sh = bias[2 * D_ + d];
    for (int k = 0; k < D_; ++k) {
        const float x = xr[k];
        const float* kp = Kk + (size_t)k * 3 * D_;
        sz += x * kp[d]; sr += x * kp[D_ + d]; sh += x * kp[2 * D_ + d];
    }
    // h = 0 -> hk = bias[1]
    const float hz = bias[3 * D_ + d], hr = bias[4 * D_ + d], hh_ = bias[5 * D_ + d];
    const float z = sigmoidf_(sz + hz);
    const float r = sigmoidf_(sr + hr);
    const float hh = tanhf(sh + r * hh_);
    h1[(size_t)row * D_ + d] = (1.f - z) * hh;   // + z*0
}

// ---------------------------------------------------------------------------
// GRU step 2: out = GRU(agg, h1).
// ---------------------------------------------------------------------------
__global__ __launch_bounds__(64) void gru2_kernel(const float* __restrict__ agg,
                                                  const float* __restrict__ h1g,
                                                  const float* __restrict__ Kk,
                                                  const float* __restrict__ Rk,
                                                  const float* __restrict__ bias,
                                                  float* __restrict__ out) {
    __shared__ float xr[D_];
    __shared__ float hrow[D_];
    const int row = blockIdx.x, d = threadIdx.x;
    xr[d]   = agg[(size_t)row * D_ + d];
    hrow[d] = h1g[(size_t)row * D_ + d];
    __syncthreads();
    float sz = bias[d],          sr = bias[D_ + d],      sh = bias[2 * D_ + d];
    float tz = bias[3 * D_ + d], tr = bias[4 * D_ + d],  th = bias[5 * D_ + d];
    for (int k = 0; k < D_; ++k) {
        const float x = xr[k], h = hrow[k];
        const float* kp = Kk + (size_t)k * 3 * D_;
        const float* rp = Rk + (size_t)k * 3 * D_;
        sz += x * kp[d]; sr += x * kp[D_ + d]; sh += x * kp[2 * D_ + d];
        tz += h * rp[d]; tr += h * rp[D_ + d]; th += h * rp[2 * D_ + d];
    }
    const float z  = sigmoidf_(sz + tz);
    const float r  = sigmoidf_(sr + tr);
    const float hh = tanhf(sh + r * th);
    const float h  = hrow[d];
    out[(size_t)row * D_ + d] = z * h + (1.f - z) * hh;
}

// ---------------------------------------------------------------------------
extern "C" void kernel_launch(void* const* d_in, const int* in_sizes, int n_in,
                              void* d_out, int out_size, void* d_ws, size_t ws_size,
                              hipStream_t stream) {
    const float* nodes = (const float*)d_in[0];
    const float* edges = (const float*)d_in[1];
    const float* mask  = (const float*)d_in[2];
    const float* W_e   = (const float*)d_in[3];
    const float* b_e   = (const float*)d_in[4];
    const float* g_k   = (const float*)d_in[5];
    const float* g_r   = (const float*)d_in[6];
    const float* g_b   = (const float*)d_in[7];
    float* out = (float*)d_out;

    char* ws = (char*)d_ws;
    float*    agg = (float*)ws;                                     // 128 KB
    float*    h1  = (float*)(ws + (size_t)B_ * N_ * D_ * 4);        // 128 KB
    _Float16* Wh  = (_Float16*)(ws + (size_t)2 * B_ * N_ * D_ * 4); // 256 KB

    prepW_kernel<<<256, 32, 0, stream>>>(W_e, Wh);
    gru1_kernel<<<B_ * N_, D_, 0, stream>>>(nodes, g_k, g_b, h1);
    msg_kernel<<<B_ * N_, 128, 0, stream>>>(nodes, edges, mask, b_e, Wh, agg);
    gru2_kernel<<<B_ * N_, D_, 0, stream>>>(agg, h1, g_k, g_r, g_b, out);
}
